// Block_10788957847731
// MI455X (gfx1250) — matrix operand order repaired and checked by reading, not compiled, not run on hardware
//
#include <hip/hip_runtime.h>
#include <math.h>

// ---------------- problem constants ----------------
#define EMBED   768
#define NBLK    8
#define BSZ     96
#define LATENT  3072     // 4*EMBED
#define HH      128
#define WW      128
#define WF      65       // W/2+1
#define NPIX    (HH*WW)  // 16384
#define NSPEC   (HH*WF)  // 8320
#define LAMBD   0.01f
#define LNEPS   1e-5f

typedef unsigned short u16;
typedef __attribute__((ext_vector_type(16))) __bf16 v16bf;
typedef __attribute__((ext_vector_type(8)))  float  v8f;

// ---------------- bf16 helpers ----------------
__device__ __forceinline__ u16 f2bf(float f) {
  unsigned u = __float_as_uint(f);
  unsigned r = u + 0x7FFFu + ((u >> 16) & 1u);   // round-to-nearest-even
  return (u16)(r >> 16);
}
__device__ __forceinline__ float bf2f(u16 h) {
  return __uint_as_float(((unsigned)h) << 16);
}

// ---------------- CDNA5 async global->LDS copies (ASYNCcnt) ----------------
__device__ __forceinline__ void async_g2l_b128(void* lds, const void* g) {
  asm volatile("global_load_async_to_lds_b128 %0, %1, off"
               :: "v"((unsigned)(uintptr_t)lds), "v"(g) : "memory");
}
__device__ __forceinline__ void async_g2l_b64(void* lds, const void* g) {
  asm volatile("global_load_async_to_lds_b64 %0, %1, off"
               :: "v"((unsigned)(uintptr_t)lds), "v"(g) : "memory");
}
__device__ __forceinline__ void async_wait0() {
  asm volatile("s_wait_asynccnt 0x0" ::: "memory");
}

// ---------------- elementwise: f32 -> bf16 (vectorized x4) ----------------
__global__ void k_f32_to_bf16(const float* __restrict__ src, u16* __restrict__ dst, long n4) {
  long i = (long)blockIdx.x * blockDim.x + threadIdx.x;
  long st = (long)gridDim.x * blockDim.x;
  for (; i < n4; i += st) {
    float4 v = reinterpret_cast<const float4*>(src)[i];
    ushort4 o;
    o.x = f2bf(v.x); o.y = f2bf(v.y); o.z = f2bf(v.z); o.w = f2bf(v.w);
    reinterpret_cast<ushort4*>(dst)[i] = o;
  }
}

// block weights (2*NBLK blocks of 96x96, stored [i][o]) -> bf16 transposed to [o][i]
__global__ void k_wblk_oc_bf16(const float* __restrict__ w, u16* __restrict__ o, long n) {
  long i = (long)blockIdx.x * blockDim.x + threadIdx.x;
  long st = (long)gridDim.x * blockDim.x;
  for (; i < n; i += st) {
    long blk = i / (BSZ*BSZ);
    int rem  = (int)(i % (BSZ*BSZ));
    int orow = rem / BSZ, k = rem % BSZ;
    o[i] = f2bf(w[blk*(BSZ*BSZ) + (long)k*BSZ + orow]);
  }
}

// ---------------- LayerNorm over last dim (C=768), one block per pixel ----------------
__global__ void k_layernorm(const float* __restrict__ x, const float* __restrict__ w,
                            const float* __restrict__ b,
                            float* __restrict__ of32, u16* __restrict__ obf, int C) {
  __shared__ float s1[256], s2[256];
  long p = blockIdx.x;
  const float* xp = x + p * C;
  float a0 = 0.f, a1 = 0.f;
  for (int c = threadIdx.x; c < C; c += 256) { float v = xp[c]; a0 += v; a1 += v*v; }
  s1[threadIdx.x] = a0; s2[threadIdx.x] = a1; __syncthreads();
  for (int s = 128; s > 0; s >>= 1) {
    if ((int)threadIdx.x < s) { s1[threadIdx.x] += s1[threadIdx.x+s]; s2[threadIdx.x] += s2[threadIdx.x+s]; }
    __syncthreads();
  }
  float mean = s1[0] / C;
  float var  = s2[0] / C - mean*mean;
  float rstd = rsqrtf(var + LNEPS);
  for (int c = threadIdx.x; c < C; c += 256) {
    float y = (xp[c] - mean) * rstd * w[c] + b[c];
    if (of32) of32[p*C + c] = y;
    if (obf)  obf [p*C + c] = f2bf(y);
  }
}

// ---------------- 128-point radix-2 FFT in LDS (128 threads) ----------------
__device__ __forceinline__ void fft128_lds(float* re, float* im, int tid, int inverse) {
  int r = __brev(tid) >> 25;                 // 7-bit reverse
  if (r > tid) {
    float tr = re[tid], ti = im[tid];
    re[tid] = re[r]; im[tid] = im[r];
    re[r] = tr;      im[r] = ti;
  }
  __syncthreads();
  const float sgn = inverse ? 1.0f : -1.0f;
  for (int s = 1; s <= 7; ++s) {
    int m = 1 << s, half = m >> 1;
    if (tid < 64) {
      int grp = tid / half, pos = tid % half;
      int i = grp * m + pos, j = i + half;
      float ang = sgn * 6.2831853071795864769f * (float)pos / (float)m;
      float wr = __cosf(ang), wi = __sinf(ang);
      float tr = wr * re[j] - wi * im[j];
      float ti = wr * im[j] + wi * re[j];
      float ur = re[i], ui = im[i];
      re[i] = ur + tr; im[i] = ui + ti;
      re[j] = ur - tr; im[j] = ui - ti;
    }
    __syncthreads();
  }
}

// forward FFT along W for real input: in [H,W,C] -> (re,im) [H,WF,C]
__global__ void k_fft_rows_fwd(const float* __restrict__ in,
                               float* __restrict__ oRe, float* __restrict__ oIm) {
  __shared__ float re[128], im[128];
  int c = blockIdx.x, h = blockIdx.y, t = threadIdx.x;
  re[t] = in[((long)h*WW + t)*EMBED + c];
  im[t] = 0.f;
  __syncthreads();
  fft128_lds(re, im, t, 0);
  if (t < WF) {
    long o = ((long)h*WF + t)*EMBED + c;
    oRe[o] = re[t]; oIm[o] = im[t];
  }
}

// complex FFT along H, in-place on planar (re,im) [H,WF,C], optional scale
__global__ void k_fft_cols(float* __restrict__ Re, float* __restrict__ Im, int inverse, float scale) {
  __shared__ float re[128], im[128];
  int c = blockIdx.x, wp = blockIdx.y, t = threadIdx.x;
  long idx = ((long)t*WF + wp)*EMBED + c;
  re[t] = Re[idx]; im[t] = Im[idx];
  __syncthreads();
  fft128_lds(re, im, t, inverse);
  Re[idx] = re[t]*scale; Im[idx] = im[t]*scale;
}

// inverse along W using Hermitian extension; out = real/128 + bias + resid
__global__ void k_ifft_rows_final(const float* __restrict__ Re, const float* __restrict__ Im,
                                  const float* __restrict__ bias, const float* __restrict__ resid,
                                  float* __restrict__ out) {
  __shared__ float re[128], im[128];
  int c = blockIdx.x, h = blockIdx.y, t = threadIdx.x;
  if (t < WF) {
    long idx = ((long)h*WF + t)*EMBED + c;
    re[t] = Re[idx]; im[t] = Im[idx];
  } else {
    int srcw = 128 - t;  // 1..63
    long idx = ((long)h*WF + srcw)*EMBED + c;
    re[t] = Re[idx]; im[t] = -Im[idx];
  }
  __syncthreads();
  fft128_lds(re, im, t, 1);
  long o = ((long)h*WW + t)*EMBED + c;
  out[o] = re[t] * (1.0f/128.0f) + bias[o] + resid[o];
}

// ---------------- generic bf16 WMMA GEMM, double-buffered async LDS staging ----------------
// out[m,n] = act( accum_prev + sgn * sum_k A[m,k]*W[n*ldw+k] + bias[n] + resid[m,n] )
// Weights (N,K) row-major, K-contiguous.  8 waves, 4x2 grid; each wave owns
// MSUB x NSUB 16x16 accumulators.  WG tile = (64*MSUB) x (32*NSUB).
// Tiles staged with GLOBAL_LOAD_ASYNC_TO_LDS (ASYNCcnt), ping-pong buffers.
#define LDSP 40
template<int MSUB, int NSUB>
__global__ void k_wmma_gemm(const u16* __restrict__ A, int lda, long blkA,
                            const u16* __restrict__ W, int ldw, long blkW,
                            float* __restrict__ outF, u16* __restrict__ outB, int ldo, long blkO,
                            const float* __restrict__ bias, long blkBias,
                            const float* __restrict__ resid,
                            int M, int N, int K, float sgn, int accum, int act) {
  constexpr int TMR = 64 * MSUB;     // A-tile rows
  constexpr int TNR = 32 * NSUB;     // B-tile rows (N)
  __shared__ alignas(16) u16 As[2][TMR * LDSP];
  __shared__ alignas(16) u16 Bs[2][TNR * LDSP];

  int z = blockIdx.z;
  A += (long)z * blkA;
  W += (long)z * blkW;
  if (outF) outF += (long)z * blkO;
  if (outB) outB += (long)z * blkO;
  if (bias) bias += (long)z * blkBias;

  int m0 = blockIdx.y * TMR;
  int n0 = blockIdx.x * TNR;
  int tid = threadIdx.x;
  int lane = tid & 31, wave = tid >> 5;
  int wm = wave & 3, wn = wave >> 2;      // 4 x 2 wave grid

  // issue async copies for one K-tile into buffer `buf`
  auto stage = [&](int buf, int k0) {
    #pragma unroll
    for (int r = 0; r < MSUB; ++r) {
      int chunk = tid + 256 * r;
      int row = chunk >> 2, k8 = (chunk & 3) * 8;
      async_g2l_b128(&As[buf][row*LDSP + k8], &A[(long)(m0 + row)*lda + k0 + k8]);
    }
    if constexpr (NSUB == 2) {
      int row = tid >> 2, k8 = (tid & 3) * 8;     // 64 rows, b128/thread
      async_g2l_b128(&Bs[buf][row*LDSP + k8], &W[(long)(n0 + row)*ldw + k0 + k8]);
    } else {
      int row = tid >> 3, k4 = (tid & 7) * 4;     // 32 rows, b64/thread
      async_g2l_b64(&Bs[buf][row*LDSP + k4], &W[(long)(n0 + row)*ldw + k0 + k4]);
    }
  };

  v8f acc[MSUB][NSUB];
  #pragma unroll
  for (int i = 0; i < MSUB; ++i)
    #pragma unroll
    for (int j = 0; j < NSUB; ++j)
      acc[i][j] = (v8f){0.f,0.f,0.f,0.f,0.f,0.f,0.f,0.f};

  stage(0, 0);
  async_wait0();
  __syncthreads();

  int cur = 0;
  for (int k0 = 0; k0 < K; k0 += 32) {
    int nxt = cur ^ 1;
    bool more = (k0 + 32) < K;
    if (more) {
      stage(nxt, k0 + 32);                       // overlap DMA with WMMA below
      __builtin_prefetch(&A[(long)(m0 + (tid >> 2))*lda + k0 + 64], 0, 3);
    }

    // fragments: A = two b128 K-slices, B = two b128 K-slices (same K order)
    int half = lane >> 4;
    union Frag { v16bf v; int4 q[2]; } af[MSUB], bf[NSUB];
    #pragma unroll
    for (int i = 0; i < MSUB; ++i) {
      int mrow = wm*16*MSUB + i*16 + (lane & 15);
      af[i].q[0] = *reinterpret_cast<const int4*>(&As[cur][mrow*LDSP + 8*half]);
      af[i].q[1] = *reinterpret_cast<const int4*>(&As[cur][mrow*LDSP + 16 + 8*half]);
    }
    #pragma unroll
    for (int j = 0; j < NSUB; ++j) {
      int nrow = wn*16*NSUB + j*16 + (lane & 15);
      bf[j].q[0] = *reinterpret_cast<const int4*>(&Bs[cur][nrow*LDSP + 8*half]);
      bf[j].q[1] = *reinterpret_cast<const int4*>(&Bs[cur][nrow*LDSP + 16 + 8*half]);
    }
    #pragma unroll
    for (int i = 0; i < MSUB; ++i)
      #pragma unroll
      for (int j = 0; j < NSUB; ++j)
        acc[i][j] = __builtin_amdgcn_wmma_f32_16x16x32_bf16(false, af[i].v, false, bf[j].v,
                                                            (short)0, acc[i][j], false, false);
    if (more) async_wait0();   // own DMA into `nxt` done; barrier makes it WG-wide
    __syncthreads();
    cur = nxt;
  }

  // ---- epilogue: D layout -> n = lane&15, m = 8*(lane>>4) + r ----
  #pragma unroll
  for (int i = 0; i < MSUB; ++i) {
    #pragma unroll
    for (int j = 0; j < NSUB; ++j) {
      int nn = n0 + wn*16*NSUB + j*16 + (lane & 15);
      int mbase = m0 + wm*16*MSUB + i*16 + (lane >> 4)*8;
      #pragma unroll
      for (int r = 0; r < 8; ++r) {
        long oi = (long)(mbase + r)*ldo + nn;
        float v = sgn * acc[i][j][r];
        if (accum && outF) v += outF[oi];
        if (bias)  v += bias[nn];
        if (resid) v += resid[oi];
        if (act == 1) v = fmaxf(v, 0.f);
        if (outF) outF[oi] = v;
        if (outB) outB[oi] = f2bf(v);
      }
    }
  }
}

// ---------------- filter modulation: complex scale/shift then relu -> bf16 ----------------
__global__ void k_filter_mod(const float* __restrict__ o1re, const float* __restrict__ o1im,
                             const float* __restrict__ mre,  const float* __restrict__ mim,
                             u16* __restrict__ ore_b, u16* __restrict__ oim_b, long n) {
  long i = (long)blockIdx.x * blockDim.x + threadIdx.x;
  long st = (long)gridDim.x * blockDim.x;
  for (; i < n; i += st) {
    long row = i / EMBED; int c = (int)(i % EMBED);
    float scre = 1.f + mre[row*1536 + c];
    float shre =        mre[row*1536 + EMBED + c];
    float scim = 1.f + mim[row*1536 + c];
    float shim =        mim[row*1536 + EMBED + c];
    float a = o1re[i], b = o1im[i];
    float nre = a*scre - b*scim + shre;
    float nim = b*scre + a*scim + shim;
    ore_b[i] = f2bf(fmaxf(nre, 0.f));
    oim_b[i] = f2bf(fmaxf(nim, 0.f));
  }
}

// ---------------- softshrink in-place on (re,im) ----------------
__global__ void k_softshrink(float* __restrict__ a, float* __restrict__ b, long n) {
  long i = (long)blockIdx.x * blockDim.x + threadIdx.x;
  long st = (long)gridDim.x * blockDim.x;
  for (; i < n; i += st) {
    float x = a[i];
    a[i] = (x > LAMBD) ? x - LAMBD : ((x < -LAMBD) ? x + LAMBD : 0.f);
    float y = b[i];
    b[i] = (y > LAMBD) ? y - LAMBD : ((y < -LAMBD) ? y + LAMBD : 0.f);
  }
}

// ---------------- MLP: h = f1*(1+scale)+shift ; gelu(exact) -> bf16 ----------------
__global__ void k_mod_gelu(const float* __restrict__ f1, const u16* __restrict__ g2,
                           u16* __restrict__ out, long n) {
  long i = (long)blockIdx.x * blockDim.x + threadIdx.x;
  long st = (long)gridDim.x * blockDim.x;
  for (; i < n; i += st) {
    long row = i / LATENT; int c = (int)(i % LATENT);
    float sc = 1.f + bf2f(g2[row*(2*LATENT) + c]);
    float sh =        bf2f(g2[row*(2*LATENT) + LATENT + c]);
    float h = f1[i]*sc + sh;
    float g = 0.5f * h * (1.f + erff(h * 0.70710678118654752440f));
    out[i] = f2bf(g);
  }
}

// ---------------- host orchestration ----------------
extern "C" void kernel_launch(void* const* d_in, const int* in_sizes, int n_in,
                              void* d_out, int out_size, void* d_ws, size_t ws_size,
                              hipStream_t stream) {
  (void)in_sizes; (void)n_in; (void)out_size; (void)ws_size;
  const float* x      = (const float*)d_in[0];
  const float* mod    = (const float*)d_in[1];
  const float* n1w    = (const float*)d_in[2];
  const float* n1b    = (const float*)d_in[3];
  const float* n2w    = (const float*)d_in[4];
  const float* n2b    = (const float*)d_in[5];
  const float* w1     = (const float*)d_in[6];   // (2,8,96,96) [i][o]
  const float* b1     = (const float*)d_in[7];   // (2,8,96)
  const float* w2     = (const float*)d_in[8];
  const float* b2     = (const float*)d_in[9];
  const float* fc1w_f = (const float*)d_in[10];  // f_c1_w (3072,768)
  const float* fc1b_f = (const float*)d_in[11];
  const float* fc2w_f = (const float*)d_in[12];  // f_c2_w (1536,3072)
  const float* fc2b_f = (const float*)d_in[13];
  const float* mlp1w  = (const float*)d_in[14];  // fc1_w (3072,768)
  const float* mlp1b  = (const float*)d_in[15];
  const float* mlp2w  = (const float*)d_in[16];  // fc2_w (768,3072)
  const float* mlp2b  = (const float*)d_in[17];
  const float* mc1w   = (const float*)d_in[18];  // (12288,768)
  const float* mc1b   = (const float*)d_in[19];
  const float* mc2w   = (const float*)d_in[20];  // (6144,12288)
  const float* mc2b   = (const float*)d_in[21];
  float* out = (float*)d_out;

  char* ws = (char*)d_ws;
  size_t cur = 0;
  auto alloc = [&](size_t bytes) -> void* {
    void* p = (void*)(ws + cur);
    cur += (bytes + 255) & ~(size_t)255;
    return p;
  };

  const long PC = (long)NPIX * EMBED;      // 16384*768
  const long SC = (long)NSPEC * EMBED;     // 8320*768
  const long WBLK = (long)NBLK*BSZ*BSZ;    // 8*96*96

  float* h_ln   = (float*)alloc(PC*4);
  float* sxre   = (float*)alloc(SC*4);
  float* sxim   = (float*)alloc(SC*4);
  float* smre   = (float*)alloc(SC*4);
  float* smim   = (float*)alloc(SC*4);
  u16* sxre_b   = (u16*)alloc(SC*2);
  u16* sxim_b   = (u16*)alloc(SC*2);
  u16* smre_b   = (u16*)alloc(SC*2);
  u16* smim_b   = (u16*)alloc(SC*2);
  u16* w1_b     = (u16*)alloc(2*WBLK*2);   // transposed to [o][i]
  u16* w2_b     = (u16*)alloc(2*WBLK*2);
  u16* fc1w_b   = (u16*)alloc((long)LATENT*EMBED*2);
  u16* fc2w_b   = (u16*)alloc((long)2*EMBED*LATENT*2);
  u16* mlp1w_b  = (u16*)alloc((long)LATENT*EMBED*2);
  u16* mlp2w_b  = (u16*)alloc((long)EMBED*LATENT*2);
  u16* mc1w_b   = (u16*)alloc((long)4*LATENT*EMBED*2);
  u16* mc2w_b   = (u16*)alloc((long)2*LATENT*4*LATENT*2);
  u16* m1re_b   = (u16*)alloc((long)NSPEC*LATENT*2);
  u16* m1im_b   = (u16*)alloc((long)NSPEC*LATENT*2);
  float* m2re   = (float*)alloc((long)NSPEC*2*EMBED*4);
  float* m2im   = (float*)alloc((long)NSPEC*2*EMBED*4);
  float* o1re   = (float*)alloc(SC*4);
  float* o1im   = (float*)alloc(SC*4);
  u16* o1re_b   = (u16*)alloc(SC*2);
  u16* o1im_b   = (u16*)alloc(SC*2);
  float* o2re   = (float*)alloc(SC*4);
  float* o2im   = (float*)alloc(SC*4);
  float* hbuf   = (float*)alloc(PC*4);
  u16* h2_b     = (u16*)alloc(PC*2);
  u16* modsp_b  = (u16*)alloc(PC*2);
  u16* g1_b     = (u16*)alloc((long)NPIX*4*LATENT*2);
  u16* g2_b     = (u16*)alloc((long)NPIX*2*LATENT*2);
  float* f1     = (float*)alloc((long)NPIX*LATENT*4);
  u16* f1_b     = (u16*)alloc((long)NPIX*LATENT*2);

  auto conv = [&](const float* s, u16* d, long n) {
    k_f32_to_bf16<<<dim3(4096), dim3(256), 0, stream>>>(s, d, n/4);
  };
  // big GEMM (tiles 128x64), all dims divide exactly
  auto gemm2 = [&](const u16* A, int lda, const u16* Wp, int ldw,
                   float* oF, u16* oB, int ldo,
                   const float* bias, const float* resid,
                   int M, int N, int K, int act) {
    dim3 g(N/64, M/128, 1);
    k_wmma_gemm<2,2><<<g, 256, 0, stream>>>(A, lda, 0, Wp, ldw, 0, oF, oB, ldo, 0,
                                            bias, 0, resid, M, N, K, 1.f, 0, act);
  };
  // block-diagonal GEMM (tiles 64x32), z = block index
  auto gemmB = [&](const u16* A, const u16* Wp, float* oF,
                   const float* bias, float sgn, int accum) {
    dim3 g(BSZ/32, NSPEC/64, NBLK);
    k_wmma_gemm<1,1><<<g, 256, 0, stream>>>(A, EMBED, BSZ, Wp, BSZ, (long)BSZ*BSZ,
                                            oF, nullptr, EMBED, BSZ,
                                            bias, BSZ, nullptr,
                                            NSPEC, BSZ, BSZ, sgn, accum, 0);
  };

  // ---- weight conversions to bf16 (block weights also transposed to [o][i]) ----
  k_wblk_oc_bf16<<<dim3(512), dim3(256), 0, stream>>>(w1, w1_b, 2*WBLK);
  k_wblk_oc_bf16<<<dim3(512), dim3(256), 0, stream>>>(w2, w2_b, 2*WBLK);
  conv(fc1w_f, fc1w_b, (long)LATENT*EMBED);
  conv(fc2w_f, fc2w_b, (long)2*EMBED*LATENT);
  conv(mlp1w, mlp1w_b, (long)LATENT*EMBED);
  conv(mlp2w, mlp2w_b, (long)EMBED*LATENT);
  conv(mc1w,  mc1w_b,  (long)4*LATENT*EMBED);
  conv(mc2w,  mc2w_b,  (long)2*LATENT*4*LATENT);

  // ---- LN1 (fp32 out for FFT) ----
  k_layernorm<<<dim3(NPIX), dim3(256), 0, stream>>>(x, n1w, n1b, h_ln, (u16*)nullptr, EMBED);

  // ---- forward rfft2 of h_ln and mod_embed (ortho: 1/128 in column pass) ----
  k_fft_rows_fwd<<<dim3(EMBED, HH), dim3(128), 0, stream>>>(h_ln, sxre, sxim);
  k_fft_cols    <<<dim3(EMBED, WF), dim3(128), 0, stream>>>(sxre, sxim, 0, 1.0f/128.0f);
  k_fft_rows_fwd<<<dim3(EMBED, HH), dim3(128), 0, stream>>>(mod, smre, smim);
  k_fft_cols    <<<dim3(EMBED, WF), dim3(128), 0, stream>>>(smre, smim, 0, 1.0f/128.0f);

  conv(sxre, sxre_b, SC); conv(sxim, sxim_b, SC);
  conv(smre, smre_b, SC); conv(smim, smim_b, SC);

  // ---- spectral modulation MLP: f_c1 (768->3072, relu) ; f_c2 (3072->1536, relu) ----
  gemm2(smre_b, EMBED, fc1w_b, EMBED, nullptr, m1re_b, LATENT, fc1b_f, nullptr,
        NSPEC, LATENT, EMBED, 1);
  gemm2(smim_b, EMBED, fc1w_b, EMBED, nullptr, m1im_b, LATENT, fc1b_f, nullptr,
        NSPEC, LATENT, EMBED, 1);
  gemm2(m1re_b, LATENT, fc2w_b, LATENT, m2re, nullptr, 2*EMBED, fc2b_f, nullptr,
        NSPEC, 2*EMBED, LATENT, 1);
  gemm2(m1im_b, LATENT, fc2w_b, LATENT, m2im, nullptr, 2*EMBED, fc2b_f, nullptr,
        NSPEC, 2*EMBED, LATENT, 1);

  // ---- block-diagonal complex matmul 1: o1 = (xr+ixi)(w1r+iw1i) + b1 ----
  const u16* w1r_b = w1_b;  const u16* w1i_b = w1_b + WBLK;
  const float* b1r = b1;    const float* b1i = b1 + NBLK*BSZ;
  gemmB(sxre_b, w1r_b, o1re, nullptr,  1.f, 0);
  gemmB(sxim_b, w1i_b, o1re, b1r,     -1.f, 1);
  gemmB(sxim_b, w1r_b, o1im, nullptr,  1.f, 0);
  gemmB(sxre_b, w1i_b, o1im, b1i,      1.f, 1);

  // ---- modulate + relu -> bf16 ----
  k_filter_mod<<<dim3(4096), dim3(256), 0, stream>>>(o1re, o1im, m2re, m2im,
                                                     o1re_b, o1im_b, SC);

  // ---- block-diagonal complex matmul 2 + b2 ----
  const u16* w2r_b = w2_b;  const u16* w2i_b = w2_b + WBLK;
  const float* b2r = b2;    const float* b2i = b2 + NBLK*BSZ;
  gemmB(o1re_b, w2r_b, o2re, nullptr,  1.f, 0);
  gemmB(o1im_b, w2i_b, o2re, b2r,     -1.f, 1);
  gemmB(o1im_b, w2r_b, o2im, nullptr,  1.f, 0);
  gemmB(o1re_b, w2i_b, o2im, b2i,      1.f, 1);

  // ---- softshrink, inverse FFT (H then W), + bias(h_ln) + residual(x) ----
  k_softshrink<<<dim3(4096), dim3(256), 0, stream>>>(o2re, o2im, SC);
  k_fft_cols  <<<dim3(EMBED, WF), dim3(128), 0, stream>>>(o2re, o2im, 1, 1.0f);
  k_ifft_rows_final<<<dim3(EMBED, HH), dim3(128), 0, stream>>>(o2re, o2im, h_ln, x, hbuf);

  // ---- LN2 -> bf16 ----
  k_layernorm<<<dim3(NPIX), dim3(256), 0, stream>>>(hbuf, n2w, n2b, (float*)nullptr, h2_b, EMBED);

  // ---- MLP modulation: m_c1 (768->12288, relu), m_c2 (12288->6144, relu) ----
  conv(mod, modsp_b, PC);
  gemm2(modsp_b, EMBED, mc1w_b, EMBED, nullptr, g1_b, 4*LATENT, mc1b, nullptr,
        NPIX, 4*LATENT, EMBED, 1);
  gemm2(g1_b, 4*LATENT, mc2w_b, 4*LATENT, nullptr, g2_b, 2*LATENT, mc2b, nullptr,
        NPIX, 2*LATENT, 4*LATENT, 1);

  // ---- fc1, modulate+gelu, fc2 (+residual into d_out) ----
  gemm2(h2_b, EMBED, mlp1w_b, EMBED, f1, nullptr, LATENT, mlp1b, nullptr,
        NPIX, LATENT, EMBED, 0);
  k_mod_gelu<<<dim3(4096), dim3(256), 0, stream>>>(f1, g2_b, f1_b, (long)NPIX*LATENT);
  gemm2(f1_b, LATENT, mlp2w_b, LATENT, out, nullptr, EMBED, mlp2b, hbuf,
        NPIX, EMBED, LATENT, 0);
}
